// LSTMNN_22127671509211
// MI455X (gfx1250) — matrix operand order repaired and checked by reading, not compiled, not run on hardware
//
#include <hip/hip_runtime.h>

// ---------------------------------------------------------------------------
// 2-layer LSTM, T=1024, B=64, I=128, H=512 — CDNA5 (gfx1250) WMMA f16 path.
//   prep kernels : fp32 -> f16 weight/x conversion, state zeroing
//   lin_in       : A0[t] = sigmoid(x @ W_in^T + b_in)  (parallel over T)
//   lstm_seq     : persistent recurrence, 32 WGs, grid barrier per step
//   lin_out      : out[t] = h1[t] @ W_out^T + b_out    (parallel over T)
// ---------------------------------------------------------------------------

typedef _Float16 h8   __attribute__((ext_vector_type(8)));
typedef _Float16 v16h __attribute__((ext_vector_type(16)));
typedef float    v8f  __attribute__((ext_vector_type(8)));

static constexpr int    T_   = 1024;
static constexpr int    Bz   = 64;
static constexpr int    Iz   = 128;
static constexpr int    Hz   = 512;
static constexpr int    NG   = 4 * Hz;          // 2048 gate columns
static constexpr int    Kc   = 2 * Hz;          // 1024 fused K (inp ++ h)
static constexpr size_t BH   = (size_t)Bz * Hz; // 32768 elements
static constexpr int    NBLK = 32;              // persistent workgroups

// ---- workspace layout (bytes, all 256-aligned) ----------------------------
static constexpr size_t OFF_BAR  = 0;
static constexpr size_t OFF_XH   = 256;                                    // [T*B, I]   f16
static constexpr size_t OFF_A0   = OFF_XH   + (size_t)T_ * Bz * Iz * 2;    // [T*B, H]   f16
static constexpr size_t OFF_H1   = OFF_A0   + (size_t)T_ * Bz * Hz * 2;    // [(T+1)*B,H]f16
static constexpr size_t OFF_W0   = OFF_H1   + (size_t)(T_ + 1) * Bz * Hz * 2; // [2048,1024] f16
static constexpr size_t OFF_W1   = OFF_W0   + (size_t)NG * Kc * 2;
static constexpr size_t OFF_WIN  = OFF_W1   + (size_t)NG * Kc * 2;         // [512,128]  f16
static constexpr size_t OFF_WOUT = OFF_WIN  + (size_t)Hz * Iz * 2;         // [128,512]  f16
static constexpr size_t OFF_BCAT = OFF_WOUT + (size_t)Iz * Hz * 2;         // [2,2048]   f32
static constexpr size_t OFF_H0   = OFF_BCAT + (size_t)2 * NG * 4;          // [2,B,H]    f16
static constexpr size_t OFF_END  = OFF_H0   + (size_t)2 * Bz * Hz * 2;

// ---- helpers --------------------------------------------------------------
#define WMMA_F16(A, Bm, C) \
    __builtin_amdgcn_wmma_f32_16x16x32_f16(false, (A), false, (Bm), (short)0, (C), false, false)

__device__ __forceinline__ v16h ld_contig16(const _Float16* __restrict__ p) {
    h8 lo = *(const h8*)p;
    h8 hi = *(const h8*)(p + 8);
    return __builtin_shufflevector(lo, hi, 0,1,2,3,4,5,6,7,8,9,10,11,12,13,14,15);
}
// A-fragment: two 8-half chunks 16 K apart (per CDNA5 16-bit A layout)
__device__ __forceinline__ v16h ld_afrag(const _Float16* __restrict__ p) {
    h8 lo = *(const h8*)p;
    h8 hi = *(const h8*)(p + 16);
    return __builtin_shufflevector(lo, hi, 0,1,2,3,4,5,6,7,8,9,10,11,12,13,14,15);
}

__device__ __forceinline__ float sigf(float x)  { return 1.0f / (1.0f + __expf(-x)); }
__device__ __forceinline__ float tanhfast(float x) { return 2.0f * sigf(2.0f * x) - 1.0f; }

// ---- prep: fp32 -> f16 conversions + zero init -----------------------------
__global__ void prep_x_kernel(const float* __restrict__ x, _Float16* __restrict__ xh) {
    size_t n = (size_t)T_ * Bz * Iz;
    size_t stride = (size_t)gridDim.x * blockDim.x;
    for (size_t i = (size_t)blockIdx.x * blockDim.x + threadIdx.x; i < n; i += stride)
        xh[i] = (_Float16)x[i];
}

__global__ void prep_w_kernel(const float* __restrict__ w_in, const float* __restrict__ w_ih,
                              const float* __restrict__ w_hh, const float* __restrict__ b_ih,
                              const float* __restrict__ b_hh, const float* __restrict__ w_out,
                              _Float16* __restrict__ W0c, _Float16* __restrict__ W1c,
                              _Float16* __restrict__ Winh, _Float16* __restrict__ Wouth,
                              float* __restrict__ bcat, _Float16* __restrict__ h0db,
                              _Float16* __restrict__ h1slot0, unsigned* __restrict__ bar) {
    size_t stride = (size_t)gridDim.x * blockDim.x;
    size_t gid = (size_t)blockIdx.x * blockDim.x + threadIdx.x;
    // fused W_cat[l] = [w_ih[l] | w_hh[l]]  -> [2048, 1024] row-major
    for (size_t i = gid; i < (size_t)NG * Kc; i += stride) {
        size_t n = i >> 10, k = i & 1023;
        size_t src = n * Hz + (k & 511);
        W0c[i] = (_Float16)((k < 512) ? w_ih[src] : w_hh[src]);
        W1c[i] = (_Float16)((k < 512) ? w_ih[(size_t)NG * Hz + src] : w_hh[(size_t)NG * Hz + src]);
    }
    for (size_t i = gid; i < (size_t)Hz * Iz; i += stride) Winh[i]  = (_Float16)w_in[i];
    for (size_t i = gid; i < (size_t)Iz * Hz; i += stride) Wouth[i] = (_Float16)w_out[i];
    for (size_t i = gid; i < (size_t)2 * NG; i += stride)  bcat[i]  = b_ih[i] + b_hh[i];
    for (size_t i = gid; i < 2 * BH; i += stride) h0db[i] = (_Float16)0.0f;
    for (size_t i = gid; i < BH; i += stride)     h1slot0[i] = (_Float16)0.0f;
    if (gid < 4) bar[gid] = 0u;
}

// ---- kernel A: A0 = sigmoid(xh @ Winh^T + b_in) ---------------------------
// M = T*B = 65536 (32 rows / block), N = 512, K = 128
__global__ __launch_bounds__(256) void lin_in_kernel(const _Float16* __restrict__ xh,
                                                     const _Float16* __restrict__ Wh,
                                                     const float* __restrict__ bin,
                                                     _Float16* __restrict__ A0) {
    const int lane = threadIdx.x & 31, wave = threadIdx.x >> 5;
    const int col = lane & 15, hi = lane >> 4;
    const size_t r0 = (size_t)blockIdx.x * 32;
    const _Float16* ap0b = xh + (r0 + col) * Iz + hi * 8;
    const _Float16* ap1b = ap0b + (size_t)16 * Iz;
    for (int nt = wave; nt < 32; nt += 8) {
        const int n = nt * 16 + col;
        const _Float16* bp = Wh + (size_t)n * Iz + hi * 8;
        const _Float16* ap0 = ap0b;
        const _Float16* ap1 = ap1b;
        v8f acc0 = {}, acc1 = {};
#pragma unroll
        for (int kt = 0; kt < 4; ++kt) {
            v16h bf = ld_afrag(bp);
            v16h a0 = ld_afrag(ap0);
            v16h a1 = ld_afrag(ap1);
            acc0 = WMMA_F16(a0, bf, acc0);
            acc1 = WMMA_F16(a1, bf, acc1);
            bp += 32; ap0 += 32; ap1 += 32;
        }
        const float bb = bin[n];
#pragma unroll
        for (int r = 0; r < 8; ++r) {
            size_t m0 = r0 + (hi ? r + 8 : r);
            A0[m0 * Hz + n]        = (_Float16)sigf(acc0[r] + bb);
            A0[(m0 + 16) * Hz + n] = (_Float16)sigf(acc1[r] + bb);
        }
    }
}

// ---- persistent recurrence ------------------------------------------------
__device__ __forceinline__ void gridbar(unsigned* bar) {
    __threadfence();
    __syncthreads();
    if (threadIdx.x == 0) {
        unsigned g = __hip_atomic_load(&bar[1], __ATOMIC_RELAXED, __HIP_MEMORY_SCOPE_AGENT);
        unsigned v = __hip_atomic_fetch_add(&bar[0], 1u, __ATOMIC_ACQ_REL, __HIP_MEMORY_SCOPE_AGENT);
        if (v == (unsigned)(NBLK - 1)) {
            __hip_atomic_store(&bar[0], 0u, __ATOMIC_RELAXED, __HIP_MEMORY_SCOPE_AGENT);
            __hip_atomic_fetch_add(&bar[1], 1u, __ATOMIC_RELEASE, __HIP_MEMORY_SCOPE_AGENT);
        } else {
            while (__hip_atomic_load(&bar[1], __ATOMIC_ACQUIRE, __HIP_MEMORY_SCOPE_AGENT) == g)
                __builtin_amdgcn_s_sleep(2);
        }
    }
    __syncthreads();
    __threadfence();
}

// One layer of one timestep: fused-K gate GEMM + LSTM cell math.
__device__ __forceinline__ void lstm_phase(const _Float16* __restrict__ As1,  // K 0..511  (input)
                                           const _Float16* __restrict__ As2,  // K 512..1023 (h_prev)
                                           const _Float16* __restrict__ Wc,   // [2048,1024] f16
                                           const float* __restrict__ bias,    // [2048]
                                           _Float16* __restrict__ Hdst,       // [B, H]
                                           float* __restrict__ ldsG,          // [4][64][16]
                                           float* __restrict__ ldsC,          // [64][16] this layer
                                           int u0, int wave, int lane) {
    const int col = lane & 15, hi = lane >> 4;
    const int gate = wave & 3, mh = wave >> 2;
    const int tileN = gate * Hz + u0;
    v8f acc0 = {}, acc1 = {};
    const _Float16* bbase = Wc + (size_t)(tileN + col) * Kc + hi * 8;
#pragma unroll
    for (int half = 0; half < 2; ++half) {
        const _Float16* As  = half ? As2 : As1;
        const _Float16* ap0 = As + (size_t)(mh * 32 + col) * Hz + hi * 8;
        const _Float16* ap1 = ap0 + (size_t)16 * Hz;
        const _Float16* bp  = bbase + (size_t)half * Hz;
#pragma unroll 4
        for (int kt = 0; kt < 16; ++kt) {
            __builtin_prefetch(bp + 256, 0, 1);  // global_prefetch_b8: weight stream
            v16h bf = ld_afrag(bp);
            v16h a0 = ld_afrag(ap0);
            v16h a1 = ld_afrag(ap1);
            acc0 = WMMA_F16(a0, bf, acc0);
            acc1 = WMMA_F16(a1, bf, acc1);
            bp += 32; ap0 += 32; ap1 += 32;
        }
    }
    // scatter 16x16 f32 tiles into the LDS gate buffer (CDNA5 C-layout)
#pragma unroll
    for (int r = 0; r < 8; ++r) {
        int b = mh * 32 + (hi ? r + 8 : r);
        ldsG[((gate * 64) + b) * 16 + col]      = acc0[r];
        ldsG[((gate * 64) + b + 16) * 16 + col] = acc1[r];
    }
    __syncthreads();
    // LSTM cell: 64 batch x 16 local units, 4 per thread
    for (int p = threadIdx.x; p < Bz * 16; p += 256) {
        int b = p >> 4, u = p & 15;
        float iv = ldsG[(0 * 64 + b) * 16 + u] + bias[u0 + u];
        float fv = ldsG[(1 * 64 + b) * 16 + u] + bias[Hz + u0 + u];
        float gv = ldsG[(2 * 64 + b) * 16 + u] + bias[2 * Hz + u0 + u];
        float ov = ldsG[(3 * 64 + b) * 16 + u] + bias[3 * Hz + u0 + u];
        float cp = ldsC[b * 16 + u];
        float cn = sigf(fv) * cp + sigf(iv) * tanhfast(gv);
        float hn = sigf(ov) * tanhfast(cn);
        ldsC[b * 16 + u] = cn;
        Hdst[(size_t)b * Hz + u0 + u] = (_Float16)hn;
    }
    __syncthreads();
}

__global__ __launch_bounds__(256, 1) void lstm_seq_kernel(const _Float16* __restrict__ A0,
                                                          const _Float16* __restrict__ W0c,
                                                          const _Float16* __restrict__ W1c,
                                                          const float* __restrict__ bcat,
                                                          _Float16* __restrict__ h0db,
                                                          _Float16* __restrict__ h1,
                                                          unsigned* __restrict__ bar) {
    __shared__ float ldsG[4 * 64 * 16];   // gate tiles     (16 KB)
    __shared__ float ldsC[2 * 64 * 16];   // persistent c    (8 KB)
    for (int i = threadIdx.x; i < 2 * 64 * 16; i += 256) ldsC[i] = 0.0f;
    __syncthreads();

    const int wg = blockIdx.x;            // 0..31 -> 16 hidden units each
    const int lane = threadIdx.x & 31;
    const int wave = threadIdx.x >> 5;
    const int u0 = wg * 16;

    for (int t = 0; t < T_; ++t) {
        const size_t rd = (size_t)(t & 1) * BH;         // h0 read slot (slot 0 zeroed)
        const size_t wr = (size_t)((t + 1) & 1) * BH;   // h0 write slot
        // layer 0: inp = A0[t], h_prev = h0[rd]
        lstm_phase(A0 + (size_t)t * BH, h0db + rd, W0c, bcat,
                   h0db + wr, ldsG, ldsC, u0, wave, lane);
        gridbar(bar);                                   // h0_new globally visible
        // layer 1: inp = h0_new, h_prev = h1[t] -> writes h1[t+1]
        lstm_phase(h0db + wr, h1 + (size_t)t * BH, W1c, bcat + NG,
                   h1 + (size_t)(t + 1) * BH, ldsG, ldsC + 64 * 16, u0, wave, lane);
        // no trailing barrier needed: h0 double-buffer + h1 history slots make
        // the next step's layer-0 writes hazard-free; next gridbar orders layer-1 reads.
    }
}

// ---- kernel C: out = h1[1..T] @ Wouth^T + b_out ---------------------------
// M = T*B = 65536 (32 rows / block), N = 128 (8 tiles, 1 per wave), K = 512
__global__ __launch_bounds__(256) void lin_out_kernel(const _Float16* __restrict__ Hv,
                                                      const _Float16* __restrict__ Wo,
                                                      const float* __restrict__ bo,
                                                      float* __restrict__ out) {
    const int lane = threadIdx.x & 31, wave = threadIdx.x >> 5;
    const int col = lane & 15, hi = lane >> 4;
    const size_t r0 = (size_t)blockIdx.x * 32;
    const int n = wave * 16 + col;
    const _Float16* bp  = Wo + (size_t)n * Hz + hi * 8;
    const _Float16* ap0 = Hv + (r0 + col) * Hz + hi * 8;
    const _Float16* ap1 = ap0 + (size_t)16 * Hz;
    v8f acc0 = {}, acc1 = {};
#pragma unroll 4
    for (int kt = 0; kt < 16; ++kt) {
        v16h bf = ld_afrag(bp);
        v16h a0 = ld_afrag(ap0);
        v16h a1 = ld_afrag(ap1);
        acc0 = WMMA_F16(a0, bf, acc0);
        acc1 = WMMA_F16(a1, bf, acc1);
        bp += 32; ap0 += 32; ap1 += 32;
    }
    const float bb = bo[n];
#pragma unroll
    for (int r = 0; r < 8; ++r) {
        size_t m0 = r0 + (hi ? r + 8 : r);
        out[m0 * Iz + n]        = acc0[r] + bb;
        out[(m0 + 16) * Iz + n] = acc1[r] + bb;
    }
}

// ---------------------------------------------------------------------------
extern "C" void kernel_launch(void* const* d_in, const int* in_sizes, int n_in,
                              void* d_out, int out_size, void* d_ws, size_t ws_size,
                              hipStream_t stream) {
    (void)in_sizes; (void)n_in; (void)out_size;
    const float* x     = (const float*)d_in[0];
    const float* w_in  = (const float*)d_in[1];
    const float* b_in  = (const float*)d_in[2];
    const float* w_ih  = (const float*)d_in[3];
    const float* w_hh  = (const float*)d_in[4];
    const float* b_ih  = (const float*)d_in[5];
    const float* b_hh  = (const float*)d_in[6];
    const float* w_out = (const float*)d_in[7];
    const float* b_out = (const float*)d_in[8];
    float* out = (float*)d_out;

    char* ws = (char*)d_ws;
    unsigned*  bar   = (unsigned*)(ws + OFF_BAR);
    _Float16*  xh    = (_Float16*)(ws + OFF_XH);
    _Float16*  A0    = (_Float16*)(ws + OFF_A0);
    _Float16*  h1    = (_Float16*)(ws + OFF_H1);
    _Float16*  W0c   = (_Float16*)(ws + OFF_W0);
    _Float16*  W1c   = (_Float16*)(ws + OFF_W1);
    _Float16*  Winh  = (_Float16*)(ws + OFF_WIN);
    _Float16*  Wouth = (_Float16*)(ws + OFF_WOUT);
    float*     bcat  = (float*)(ws + OFF_BCAT);
    _Float16*  h0db  = (_Float16*)(ws + OFF_H0);
    if (ws_size < OFF_END) return;  // workspace too small; nothing sane to do

    prep_x_kernel<<<4096, 256, 0, stream>>>(x, xh);
    prep_w_kernel<<<2048, 256, 0, stream>>>(w_in, w_ih, w_hh, b_ih, b_hh, w_out,
                                            W0c, W1c, Winh, Wouth, bcat, h0db, h1, bar);
    lin_in_kernel<<<2048, 256, 0, stream>>>(xh, Winh, b_in, A0);
    lstm_seq_kernel<<<NBLK, 256, 0, stream>>>(A0, W0c, W1c, bcat, h0db, h1, bar);
    lin_out_kernel<<<2048, 256, 0, stream>>>(h1 + BH, Wouth, b_out, out);
}
